// Neuralsymbol_58892591563455
// MI455X (gfx1250) — compile-verified, matrix operand order checked by reading, not run on hardware
//
#include <hip/hip_runtime.h>
#include <cmath>

// Problem sizes (from the reference)
static constexpr int NV   = 512;   // V
static constexpr int NH   = 1024;  // H
static constexpr int NB   = 128;   // B
static constexpr int NT   = 128;   // T
static constexpr int NSTK = 4;     // stack depth

// ---------------- WMMA plumbing (CDNA5, wave32) ----------------
typedef __attribute__((ext_vector_type(16))) __bf16 v16bf;
typedef __attribute__((ext_vector_type(8)))  __bf16 bf16x8;
typedef __attribute__((ext_vector_type(8)))  float  v8f;

union FragU { v16bf v; bf16x8 h[2]; };

// Load a 16x32 (A) or 32x16-as-rows (B) bf16 fragment from a row-major matrix.
// ISA 7.12.2: lane half selects which K-halves the lane holds; two 16B chunks.
__device__ __forceinline__ v16bf load_frag(const __bf16* base, int ld, int lane, int k0) {
  const int row  = lane & 15;
  const int half = lane >> 4;
  const __bf16* p = base + (size_t)row * ld + k0 + half * 8;
  FragU u;
  u.h[0] = *(const bf16x8*)(p);
  u.h[1] = *(const bf16x8*)(p + 16);
  return u.v;
}

// C = A(16xK) * W^T, W is N x K row-major bf16.  Y is 16 x N f32 scratch.
// 8 waves split the N/16 tiles; each wave's lanes are fully active (EXEC all-1s).
__device__ __forceinline__ void gemm16(const __bf16* A, int lda,
                                       const __bf16* W, int ldb,
                                       float* Y, int N, int K,
                                       int wave, int lane) {
  const int ntiles = N >> 4;
  for (int nt = wave; nt < ntiles; nt += 8) {
    v8f acc = {};
    const __bf16* Wb = W + (size_t)(nt << 4) * ldb;
    for (int k0 = 0; k0 < K; k0 += 32) {
      v16bf a = load_frag(A,  lda, lane, k0);
      v16bf b = load_frag(Wb, ldb, lane, k0);
      acc = __builtin_amdgcn_wmma_f32_16x16x32_bf16(false, a, false, b,
                                                    (short)0, acc, false, false);
    }
    const int n  = (nt << 4) + (lane & 15);
    const int m0 = (lane >> 4) << 3;
#pragma unroll
    for (int r = 0; r < 8; ++r) Y[(size_t)(m0 + r) * N + n] = acc[r];
  }
}

// ---------------- prep: row-normalize (nl folds) + bf16 convert ----------------
__global__ __launch_bounds__(256) void prep_rows_bf16(const float* __restrict__ src,
                                                      __bf16* __restrict__ dst,
                                                      int cols, int do_norm) {
  __shared__ float red[256];
  const int row = blockIdx.x;
  const float* s = src + (size_t)row * cols;
  __bf16* d = dst + (size_t)row * cols;
  float ss = 0.f;
  for (int c = threadIdx.x; c < cols; c += 256) { float v = s[c]; ss += v * v; }
  red[threadIdx.x] = ss;
  __syncthreads();
  for (int k = 128; k > 0; k >>= 1) {
    if (threadIdx.x < k) red[threadIdx.x] += red[threadIdx.x + k];
    __syncthreads();
  }
  const float inv = do_norm ? (1.0f / fmaxf(sqrtf(red[0]), 1e-12f)) : 1.0f;
  for (int c = threadIdx.x; c < cols; c += 256) d[c] = (__bf16)(s[c] * inv);
}

__global__ __launch_bounds__(256) void prep_rows_f32(const float* __restrict__ src,
                                                     float* __restrict__ dst,
                                                     int cols, int do_norm) {
  __shared__ float red[256];
  const int row = blockIdx.x;
  const float* s = src + (size_t)row * cols;
  float* d = dst + (size_t)row * cols;
  float ss = 0.f;
  for (int c = threadIdx.x; c < cols; c += 256) { float v = s[c]; ss += v * v; }
  red[threadIdx.x] = ss;
  __syncthreads();
  for (int k = 128; k > 0; k >>= 1) {
    if (threadIdx.x < k) red[threadIdx.x] += red[threadIdx.x + k];
    __syncthreads();
  }
  const float inv = do_norm ? (1.0f / fmaxf(sqrtf(red[0]), 1e-12f)) : 1.0f;
  for (int c = threadIdx.x; c < cols; c += 256) d[c] = s[c] * inv;
}

// ---------------- persistent scan kernel: 8 WGs x 256 threads ----------------
// Each workgroup owns batch rows [16*blockIdx.x, +16) for all T steps.
__global__ __launch_bounds__(256) void step_all(
    const float* __restrict__ x,
    const float* __restrict__ cop_b1, const float* __restrict__ cop_W2, const float* __restrict__ cop_b2,
    const float* __restrict__ wop_b1, const float* __restrict__ wop_W2, const float* __restrict__ wop_b2,
    const float* __restrict__ sel_W2, const float* __restrict__ out_sym,
    const float* __restrict__ stack_init,
    const float* __restrict__ ctrl_sharp, const float* __restrict__ work_sharp,
    const __bf16* __restrict__ nCtrl1, const __bf16* __restrict__ nCtrl2,
    const __bf16* __restrict__ nWork1, const __bf16* __restrict__ nWork2,
    const __bf16* __restrict__ nSel1,
    const __bf16* __restrict__ bCop1, const __bf16* __restrict__ bWop1,
    const float* __restrict__ nIsin1, const float* __restrict__ nIsin2,
    __bf16* __restrict__ gWin, __bf16* __restrict__ gCwt,
    float* __restrict__ cS, float* __restrict__ wS,
    float* __restrict__ gWork, float* __restrict__ gNewC, float* __restrict__ gNewW,
    float* __restrict__ gY_all,
    float* __restrict__ out)
{
  __shared__ __align__(16) __bf16 s_xy[16 * NV];   // 16KB: xy (A for gates/ctrl)
  __shared__ __align__(16) __bf16 s_h[16 * NH];    // 32KB: hidden (A for layer 2)
  __shared__ float s_red[16][16][5];
  __shared__ float s_cp[16][4], s_wp[16][4];       // stack pointers persist across t
  __shared__ float s_r[16][4];
  __shared__ float s_nxy[16], s_nwin[16], s_ncwt[16], s_nh[16];
  __shared__ float s_rowmax[16];
  __shared__ float s_gate[6][16];                  // cop push/pop/null, wop push/pop/null
  __shared__ float s_sel[16][5];

  const int tid  = threadIdx.x;
  const int wave = tid >> 5;        // wave32
  const int lane = tid & 31;
  const int m16  = tid >> 4;        // 16 threads per batch row for vector work
  const int l16  = tid & 15;
  const int b0   = blockIdx.x * 16;
  float* Y = gY_all + (size_t)blockIdx.x * 16 * NH;

  // ---- init: stacks = ZERO_OFF, hard-push init vector (slot 1), p=[0,1,0,0]
  for (int e = tid; e < 16 * NSTK * NV; e += 256) {
    const int m = e >> 11, n = (e >> 9) & 3, v = e & 511;
    const float val = (n == 1) ? stack_init[v] : 1e-6f;
    const size_t idx = ((size_t)(b0 + m) * NSTK + n) * NV + v;
    cS[idx] = val; wS[idx] = val;
  }
  if (tid < 64) {
    const int m = tid >> 2, n = tid & 3;
    const float pv = (n == 1) ? 1.f : 0.f;
    s_cp[m][n] = pv; s_wp[m][n] = pv;
  }
  const float sharpC = ctrl_sharp[0];
  const float sharpW = work_sharp[0];
  __syncthreads();

  for (int t = 0; t < NT; ++t) {
    const float* xr = x + ((size_t)(b0 + m16) * NT + t) * NV;

    // ===== phase 1a: typ row stats: ||inp||, inp . nW_isin1[i]
    {
      float ss = 0.f, d0 = 0.f, d1 = 0.f, d2 = 0.f, d3 = 0.f;
      for (int v = l16; v < NV; v += 16) {
        const float xv = xr[v];
        ss += xv * xv;
        d0 += xv * nIsin1[0 * NV + v];
        d1 += xv * nIsin1[1 * NV + v];
        d2 += xv * nIsin1[2 * NV + v];
        d3 += xv * nIsin1[3 * NV + v];
      }
      s_red[m16][l16][0] = ss; s_red[m16][l16][1] = d0; s_red[m16][l16][2] = d1;
      s_red[m16][l16][3] = d2; s_red[m16][l16][4] = d3;
    }
    __syncthreads();
    if (tid < 16) {
      float ss = 0.f, d[4] = {0.f, 0.f, 0.f, 0.f};
      for (int l = 0; l < 16; ++l) {
        ss   += s_red[tid][l][0];
        d[0] += s_red[tid][l][1]; d[1] += s_red[tid][l][2];
        d[2] += s_red[tid][l][3]; d[3] += s_red[tid][l][4];
      }
      const float inv = 1.0f / fmaxf(sqrtf(ss), 1e-12f);
      float tt[4], mx = 0.f;
      for (int i = 0; i < 4; ++i) { tt[i] = fmaxf(d[i] * inv, 0.f); mx = fmaxf(mx, tt[i]); }
      float e[4], se = 0.f;
      for (int i = 0; i < 4; ++i) { e[i] = expf(tt[i] - mx); se += e[i]; }
      float rr = 0.f;
      for (int i = 0; i < 4; ++i) { e[i] /= se; rr += e[i] * e[i]; }
      const float rinv = 1.0f / fmaxf(sqrtf(rr), 1e-12f);
      for (int i = 0; i < 4; ++i) s_r[tid][i] = e[i] * rinv;
    }
    __syncthreads();

    // ===== phase 1b: pop reads, typ, xy / win / cwt (bf16 A-matrices) + norms
    {
      const float cp0 = s_cp[m16][0], cp1 = s_cp[m16][1], cp2 = s_cp[m16][2], cp3 = s_cp[m16][3];
      const float wq0 = s_wp[m16][0], wq1 = s_wp[m16][1], wq2 = s_wp[m16][2], wq3 = s_wp[m16][3];
      const float r0 = s_r[m16][0], r1 = s_r[m16][1], r2 = s_r[m16][2], r3 = s_r[m16][3];
      const float* cSm = cS + (size_t)(b0 + m16) * NSTK * NV;
      const float* wSm = wS + (size_t)(b0 + m16) * NSTK * NV;
      const size_t rowo = (size_t)(b0 + m16) * NV;
      float axy = 0.f, awin = 0.f, acwt = 0.f;
      for (int v = l16; v < NV; v += 16) {
        const float inp = xr[v];
        const float ctl = cSm[v] * cp0 + cSm[NV + v] * cp1 + cSm[2 * NV + v] * cp2 + cSm[3 * NV + v] * cp3;
        const float wk  = wSm[v] * wq0 + wSm[NV + v] * wq1 + wSm[2 * NV + v] * wq2 + wSm[3 * NV + v] * wq3;
        const float ty  = tanhf(r0 * nIsin2[v * 4 + 0] + r1 * nIsin2[v * 4 + 1] +
                                r2 * nIsin2[v * 4 + 2] + r3 * nIsin2[v * 4 + 3]);
        const float xyv = ctl * ty;
        const float wiv = wk * inp * ty;
        const float cwv = ctl * wk * ty;
        s_xy[m16 * NV + v] = (__bf16)xyv;
        gWin[rowo + v] = (__bf16)wiv;
        gCwt[rowo + v] = (__bf16)cwv;
        gWork[rowo + v] = wk;
        axy += xyv * xyv; awin += wiv * wiv; acwt += cwv * cwv;
      }
      s_red[m16][l16][0] = axy; s_red[m16][l16][1] = awin; s_red[m16][l16][2] = acwt;
    }
    __syncthreads();
    float tpc = 0.f, tpw = 0.f;
    if (tid < 16) {
      float a = 0.f, b = 0.f, c = 0.f;
      for (int l = 0; l < 16; ++l) { a += s_red[tid][l][0]; b += s_red[tid][l][1]; c += s_red[tid][l][2]; }
      s_nxy[tid]  = fmaxf(sqrtf(a), 1e-12f);
      s_nwin[tid] = fmaxf(sqrtf(b), 1e-12f);
      s_ncwt[tid] = fmaxf(sqrtf(c), 1e-12f);
    }
    if (tid < 64) {  // pop: p <- roll(p, -1)
      const int pm = tid >> 2, pn = tid & 3;
      tpc = s_cp[pm][(pn + 1) & 3];
      tpw = s_wp[pm][(pn + 1) & 3];
    }
    __syncthreads();
    if (tid < 64) {
      const int pm = tid >> 2, pn = tid & 3;
      s_cp[pm][pn] = tpc; s_wp[pm][pn] = tpw;
    }
    __syncthreads();

    // ===== phase 2: 6 gate heads: sigmoid( softmax(xy@W1^T + b1) . W2 + b2 )
    for (int g = 0; g < 6; ++g) {
      const __bf16* W1 = (g < 3) ? (bCop1 + (size_t)g * NH * NV)
                                 : (bWop1 + (size_t)(g - 3) * NH * NV);
      gemm16(s_xy, NV, W1, NV, Y, NH, NV, wave, lane);
      __syncthreads();
      const float* b1 = (g < 3) ? (cop_b1 + g * NH) : (wop_b1 + (g - 3) * NH);
      const float* W2 = (g < 3) ? (cop_W2 + g * NH) : (wop_W2 + (g - 3) * NH);
      const float b2  = (g < 3) ? cop_b2[g] : wop_b2[g - 3];
      {
        float mx = -1e30f;
        for (int h = l16; h < NH; h += 16) mx = fmaxf(mx, Y[m16 * NH + h] + b1[h]);
        s_red[m16][l16][0] = mx;
      }
      __syncthreads();
      if (tid < 16) {
        float mx = -1e30f;
        for (int l = 0; l < 16; ++l) mx = fmaxf(mx, s_red[tid][l][0]);
        s_rowmax[tid] = mx;
      }
      __syncthreads();
      {
        const float mx = s_rowmax[m16];
        float se = 0.f, sd = 0.f;
        for (int h = l16; h < NH; h += 16) {
          const float e = expf(Y[m16 * NH + h] + b1[h] - mx);
          se += e; sd += e * W2[h];
        }
        s_red[m16][l16][0] = se; s_red[m16][l16][1] = sd;
      }
      __syncthreads();
      if (tid < 16) {
        float se = 0.f, sd = 0.f;
        for (int l = 0; l < 16; ++l) { se += s_red[tid][l][0]; sd += s_red[tid][l][1]; }
        const float z = sd / se + b2;
        s_gate[g][tid] = 1.0f / (1.0f + expf(-z));
      }
      __syncthreads();
    }

    // ===== phase 3: ctrl / work two-layer nl MLPs (norms folded into epilogues)
    for (int pipe = 0; pipe < 2; ++pipe) {
      const __bf16* A1  = (pipe == 0) ? (const __bf16*)s_xy : (gWin + (size_t)b0 * NV);
      const __bf16* Wa  = (pipe == 0) ? nCtrl1 : nWork1;
      const __bf16* Wb2 = (pipe == 0) ? nCtrl2 : nWork2;
      const float* nrm  = (pipe == 0) ? s_nxy : s_nwin;
      float* dst        = (pipe == 0) ? gNewC : gNewW;
      gemm16(A1, NV, Wa, NV, Y, NH, NV, wave, lane);
      __syncthreads();
      {
        const float inv = 1.0f / nrm[m16];
        float ss = 0.f;
        for (int h = l16; h < NH; h += 16) {
          const float hc = fmaxf(Y[m16 * NH + h] * inv, 0.f);  // relu(nl(.))
          s_h[m16 * NH + h] = (__bf16)hc;
          ss += hc * hc;
        }
        s_red[m16][l16][0] = ss;
      }
      __syncthreads();
      if (tid < 16) {
        float ss = 0.f;
        for (int l = 0; l < 16; ++l) ss += s_red[tid][l][0];
        s_nh[tid] = fmaxf(sqrtf(ss), 1e-12f);
      }
      __syncthreads();
      gemm16(s_h, NH, Wb2, NH, Y, NV, NH, wave, lane);
      __syncthreads();
      {
        const float inv = 1.0f / s_nh[m16];
        float* drow = dst + (size_t)(b0 + m16) * NV;
        for (int v = l16; v < NV; v += 16) drow[v] = tanhf(Y[m16 * NV + v] * inv);
      }
      __syncthreads();
    }

    // ===== phase 4: sel head + output row
    gemm16(gCwt + (size_t)b0 * NV, NV, nSel1, NV, Y, NH, NV, wave, lane);
    __syncthreads();
    {
      const float inv = 1.0f / s_ncwt[m16];
      float d[5] = {0.f, 0.f, 0.f, 0.f, 0.f};
      for (int h = l16; h < NH; h += 16) {
        const float sv = fmaxf(Y[m16 * NH + h] * inv, 0.f);
#pragma unroll
        for (int k = 0; k < 5; ++k) d[k] += sv * sel_W2[k * NH + h];
      }
#pragma unroll
      for (int k = 0; k < 5; ++k) s_red[m16][l16][k] = d[k];
    }
    __syncthreads();
    if (tid < 16) {
      float d[5] = {0.f, 0.f, 0.f, 0.f, 0.f};
      for (int l = 0; l < 16; ++l)
        for (int k = 0; k < 5; ++k) d[k] += s_red[tid][l][k];
      float mx = d[0];
      for (int k = 1; k < 5; ++k) mx = fmaxf(mx, d[k]);
      float se = 0.f;
      for (int k = 0; k < 5; ++k) { d[k] = expf(d[k] - mx); se += d[k]; }
      for (int k = 0; k < 5; ++k) s_sel[tid][k] = d[k] / se;
    }
    __syncthreads();
    {
      const float s0 = s_sel[m16][0], s1 = s_sel[m16][1], s2 = s_sel[m16][2],
                  s3 = s_sel[m16][3], s4 = s_sel[m16][4];
      const float* wrow = gWork + (size_t)(b0 + m16) * NV;
      float* orow = out + ((size_t)(b0 + m16) * NT + t) * NV;
      for (int v = l16; v < NV; v += 16) {
        orow[v] = s0 * wrow[v] + s1 * out_sym[v] + s2 * out_sym[NV + v] +
                  s3 * out_sym[2 * NV + v] + s4 * out_sym[3 * NV + v];
      }
    }

    // ===== phase 5: soft stack updates (contents, then sharpened pointers)
    {
      const float pushC = s_gate[0][m16], popC = s_gate[1][m16], nopC = s_gate[2][m16];
      const float pushW = s_gate[3][m16], popW = s_gate[4][m16], nopW = s_gate[5][m16];
      const float keepC = popC + nopC, keepW = popW + nopW;
      float ppc[4], ppw[4];
#pragma unroll
      for (int n = 0; n < 4; ++n) { ppc[n] = s_cp[m16][(n + 3) & 3]; ppw[n] = s_wp[m16][(n + 3) & 3]; }
      float* cSm = cS + (size_t)(b0 + m16) * NSTK * NV;
      float* wSm = wS + (size_t)(b0 + m16) * NSTK * NV;
      const float* vc = gNewC + (size_t)(b0 + m16) * NV;
      const float* vw = gNewW + (size_t)(b0 + m16) * NV;
#pragma unroll
      for (int n = 0; n < 4; ++n) {
        for (int v = l16; v < NV; v += 16) {
          float so = cSm[n * NV + v];
          cSm[n * NV + v] = pushC * (so * (1.f - ppc[n]) + vc[v] * ppc[n]) + keepC * so;
          so = wSm[n * NV + v];
          wSm[n * NV + v] = pushW * (so * (1.f - ppw[n]) + vw[v] * ppw[n]) + keepW * so;
        }
      }
    }
    __syncthreads();
    float npc = 0.f, npw = 0.f;
    if (tid < 64) {
      const int pm = tid >> 2, pn = tid & 3;
      const float pC = s_cp[pm][pn], pCm = s_cp[pm][(pn + 3) & 3], pCp = s_cp[pm][(pn + 1) & 3];
      const float pW = s_wp[pm][pn], pWm = s_wp[pm][(pn + 3) & 3], pWp = s_wp[pm][(pn + 1) & 3];
      npc = s_gate[0][pm] * pCm + s_gate[1][pm] * pCp + s_gate[2][pm] * pC;
      npw = s_gate[3][pm] * pWm + s_gate[4][pm] * pWp + s_gate[5][pm] * pW;
      npc = powf(fmaxf(npc, 0.f), sharpC);
      npw = powf(fmaxf(npw, 0.f), sharpW);
    }
    __syncthreads();
    if (tid < 64) {
      const int pm = tid >> 2, pn = tid & 3;
      s_cp[pm][pn] = npc; s_wp[pm][pn] = npw;
    }
    __syncthreads();
    if (tid < 16) {
      float sc = 0.f, sw = 0.f;
      for (int n = 0; n < 4; ++n) { sc += s_cp[tid][n]; sw += s_wp[tid][n]; }
      const float ic = 1.0f / (sc + 1e-12f), iw = 1.0f / (sw + 1e-12f);
      for (int n = 0; n < 4; ++n) { s_cp[tid][n] *= ic; s_wp[tid][n] *= iw; }
    }
    __syncthreads();
  } // t
}

// ---------------- host-side launch ----------------
extern "C" void kernel_launch(void* const* d_in, const int* in_sizes, int n_in,
                              void* d_out, int out_size, void* d_ws, size_t ws_size,
                              hipStream_t stream) {
  (void)in_sizes; (void)n_in; (void)out_size; (void)ws_size;
  const float* x          = (const float*)d_in[0];
  const float* W_isin1    = (const float*)d_in[1];
  const float* W_isin2    = (const float*)d_in[2];
  const float* cop_W1     = (const float*)d_in[3];
  const float* cop_b1     = (const float*)d_in[4];
  const float* cop_W2     = (const float*)d_in[5];
  const float* cop_b2     = (const float*)d_in[6];
  const float* wop_W1     = (const float*)d_in[7];
  const float* wop_b1     = (const float*)d_in[8];
  const float* wop_W2     = (const float*)d_in[9];
  const float* wop_b2     = (const float*)d_in[10];
  const float* ctrl_W1    = (const float*)d_in[11];
  const float* ctrl_W2    = (const float*)d_in[12];
  const float* work_W1    = (const float*)d_in[13];
  const float* work_W2    = (const float*)d_in[14];
  const float* sel_W1     = (const float*)d_in[15];
  const float* sel_W2     = (const float*)d_in[16];
  const float* out_sym    = (const float*)d_in[17];
  const float* stack_init = (const float*)d_in[18];
  const float* c_sharp    = (const float*)d_in[19];
  const float* w_sharp    = (const float*)d_in[20];

  // Workspace layout (~16 MB total; fits comfortably in the 192 MB L2).
  char* ws = (char*)d_ws;
  size_t off = 0;
  auto take = [&](size_t bytes) -> char* {
    char* p = ws + off;
    off = (off + bytes + 255) & ~(size_t)255;
    return p;
  };
  const size_t HV = (size_t)NH * NV;
  __bf16* nCtrl1 = (__bf16*)take(HV * 2);
  __bf16* nCtrl2 = (__bf16*)take(HV * 2);
  __bf16* nWork1 = (__bf16*)take(HV * 2);
  __bf16* nWork2 = (__bf16*)take(HV * 2);
  __bf16* nSel1  = (__bf16*)take(HV * 2);
  __bf16* bCop1  = (__bf16*)take(3 * HV * 2);
  __bf16* bWop1  = (__bf16*)take(3 * HV * 2);
  __bf16* gWin   = (__bf16*)take((size_t)NB * NV * 2);
  __bf16* gCwt   = (__bf16*)take((size_t)NB * NV * 2);
  float* nIsin1  = (float*)take((size_t)4 * NV * 4);
  float* nIsin2  = (float*)take((size_t)NV * 4 * 4);
  float* cS      = (float*)take((size_t)NB * NSTK * NV * 4);
  float* wS      = (float*)take((size_t)NB * NSTK * NV * 4);
  float* gWork   = (float*)take((size_t)NB * NV * 4);
  float* gNewC   = (float*)take((size_t)NB * NV * 4);
  float* gNewW   = (float*)take((size_t)NB * NV * 4);
  float* gY      = (float*)take((size_t)8 * 16 * NH * 4);
  float* out     = (float*)d_out;

  // Prep: nl() weight-row norms are time-invariant -> normalize once, cast bf16.
  prep_rows_bf16<<<NH,     256, 0, stream>>>(ctrl_W1, nCtrl1, NV, 1);
  prep_rows_bf16<<<NV,     256, 0, stream>>>(ctrl_W2, nCtrl2, NH, 1);
  prep_rows_bf16<<<NH,     256, 0, stream>>>(work_W1, nWork1, NV, 1);
  prep_rows_bf16<<<NV,     256, 0, stream>>>(work_W2, nWork2, NH, 1);
  prep_rows_bf16<<<NH,     256, 0, stream>>>(sel_W1,  nSel1,  NV, 1);
  prep_rows_bf16<<<3 * NH, 256, 0, stream>>>(cop_W1,  bCop1,  NV, 0);  // gates: plain linear
  prep_rows_bf16<<<3 * NH, 256, 0, stream>>>(wop_W1,  bWop1,  NV, 0);
  prep_rows_f32 <<<4,      256, 0, stream>>>(W_isin1, nIsin1, NV, 1);
  prep_rows_f32 <<<NV,     256, 0, stream>>>(W_isin2, nIsin2, 4,  1);

  // Persistent scan: 8 independent batch tiles, full T recurrence, no global sync.
  step_all<<<8, 256, 0, stream>>>(
      x, cop_b1, cop_W2, cop_b2, wop_b1, wop_W2, wop_b2,
      sel_W2, out_sym, stack_init, c_sharp, w_sharp,
      nCtrl1, nCtrl2, nWork1, nWork2, nSel1, bCop1, bWop1,
      nIsin1, nIsin2, gWin, gCwt, cS, wS, gWork, gNewC, gNewW, gY, out);
}
